// ModifiedDGCNN_37692632990206
// MI455X (gfx1250) — matrix operand order repaired and checked by reading, not compiled, not run on hardware
//
#include <hip/hip_runtime.h>
#include <hip/hip_bf16.h>

typedef __attribute__((ext_vector_type(16))) _Float16 v16h;
typedef __attribute__((ext_vector_type(8)))  _Float16 v8h;
typedef __attribute__((ext_vector_type(8)))  float    v8f;

union H16 { v16h v; v8h h[2]; };
union F8  { v8f v; float f[8]; };

// ---- WMMA operand gathers (layouts per CDNA5 ISA 7.12.2) ----
// A (16x32 f16): lane<16 -> row lp, K = kb+0..7 and kb+16..23 ; lane>=16 -> K = kb+8..15 and kb+24..31
__device__ inline v16h load_a_chunk(const _Float16* __restrict__ Xrow, int kb, int half) {
  H16 u;
  u.h[0] = *(const v8h*)(Xrow + kb + half * 8);
  u.h[1] = *(const v8h*)(Xrow + kb + 16 + half * 8);
  return u.v;
}
// B (32x16 f16): lane<16 -> col lp, K = kb+0..7 and kb+16..23 ; lane>=16 -> K = kb+8..15 and kb+24..31 (row of the K-major/pre-transposed matrix)
__device__ inline v16h load_b_chunk(const _Float16* __restrict__ Wrow, int kb, int half) {
  H16 u;
  u.h[0] = *(const v8h*)(Wrow + kb + half * 8);
  u.h[1] = *(const v8h*)(Wrow + kb + 16 + half * 8);
  return u.v;
}

// ================= input prep: split x into padded-f16 xyz / geom, sq norms =================
__global__ void prep_inputs_kernel(const float* __restrict__ x, _Float16* __restrict__ xyzh,
                                   _Float16* __restrict__ geomh, float* __restrict__ sq0) {
  int n = blockIdx.x * 256 + threadIdx.x;
  const float* r = x + (size_t)n * 9;
  float s = 0.f;
  for (int k = 0; k < 32; k++) {
    float v = (k < 3) ? r[k] : 0.f;
    float g = (k < 6) ? r[3 + k] : 0.f;
    xyzh[(size_t)n * 32 + k] = (_Float16)v;
    geomh[(size_t)n * 32 + k] = (_Float16)g;
    if (k < 3) s += v * v;
  }
  sq0[n] = s;
}

// ============ weight prep: wt[o*Kpad+k] = w[rowoff+k][o] (- w[diffoff+k][o]) , f16, K padded ============
__global__ void prep_wt_kernel(const float* __restrict__ w, _Float16* __restrict__ wt,
                               int Kin, int O, int Kpad, int rowoff, int diffoff) {
  size_t i = (size_t)blockIdx.x * 256 + threadIdx.x;
  if (i >= (size_t)O * Kpad) return;
  int k = (int)(i % Kpad);
  int o = (int)(i / Kpad);
  float v = 0.f;
  if (k < Kin) {
    v = w[(size_t)(rowoff + k) * O + o];
    if (diffoff >= 0) v -= w[(size_t)(diffoff + k) * O + o];
  }
  wt[i] = (_Float16)v;
}

// ================= generic WMMA GEMM: Y = X(Nn x Kpad,f16) * Wt^T(O x Kpad,f16) + bias =================
__global__ __launch_bounds__(256) void gemm_kernel(const _Float16* __restrict__ X,
                                                   const _Float16* __restrict__ Wt,
                                                   const float* __restrict__ bias,
                                                   float* __restrict__ Y, _Float16* __restrict__ Yh,
                                                   int Kpad, int O, int relu) {
  int wid = threadIdx.x >> 5, lane = threadIdx.x & 31;
  int half = lane >> 4, lp = lane & 15;
  int tile = blockIdx.x * 8 + wid;
  int otiles = O >> 4;
  int rb = (tile / otiles) << 4;
  int cb = (tile % otiles) << 4;
  const _Float16* Xrow = X + (size_t)(rb + lp) * Kpad;
  const _Float16* Wrow = Wt + (size_t)(cb + lp) * Kpad;
  v8f c = {};
  for (int kb = 0; kb < Kpad; kb += 32) {
    v16h a = load_a_chunk(Xrow, kb, half);
    v16h b = load_b_chunk(Wrow, kb, half);
    c = __builtin_amdgcn_wmma_f32_16x16x32_f16(false, a, false, b, (short)0, c, false, false);
  }
  int col = cb + lp;
  float bv = bias ? bias[col] : 0.f;
  F8 u; u.v = c;
  for (int j = 0; j < 8; j++) {
    int row = rb + j + half * 8;
    float y = u.f[j] + bv;
    if (relu) y = y > 0.f ? y : 0.f;
    if (Y)  Y[(size_t)row * O + col] = y;
    if (Yh) Yh[(size_t)row * O + col] = (_Float16)y;
  }
}

// ================= kNN: WMMA Gram -> dist tiles in LDS -> chunked top-20 merge =================
template <int KP>
__global__ __launch_bounds__(256) void knn_kernel(const _Float16* __restrict__ Xh,
                                                  const float* __restrict__ sq,
                                                  int* __restrict__ idxout, int N) {
  constexpr int C = 512;
  __shared__ float sdist[16][C + 20];
  __shared__ float srunv[16][20];
  __shared__ int   sruni[16][20];
  __shared__ float snewv[16][20];
  __shared__ int   snewi[16][20];

  int nt = N >> 4;
  int b = blockIdx.x / nt;
  int rb = (blockIdx.x % nt) << 4;
  int wid = threadIdx.x >> 5, lane = threadIdx.x & 31;
  int half = lane >> 4, lp = lane & 15;
  const _Float16* Xb = Xh + (size_t)b * N * KP;
  const float* sqb = sq + (size_t)b * N;

  if (threadIdx.x < 16 * 20) {
    ((float*)srunv)[threadIdx.x] = 3.0e38f;
    ((int*)sruni)[threadIdx.x] = 0;
  }

  // hoist the 16 query rows (A operand) into registers
  v16h a[KP / 32];
  {
    const _Float16* Xrow = Xb + (size_t)(rb + lp) * KP;
#pragma unroll
    for (int ch = 0; ch < KP / 32; ch++) a[ch] = load_a_chunk(Xrow, ch * 32, half);
  }
  float sqi[8];
#pragma unroll
  for (int j = 0; j < 8; j++) sqi[j] = sqb[rb + j + half * 8];
  __syncthreads();

  for (int cc = 0; cc < N; cc += C) {
    // ---- phase 1: distance tile for this 512-column chunk ----
    for (int t = wid; t < (C >> 4); t += 8) {
      int cb = cc + (t << 4);
      const _Float16* Brow = Xb + (size_t)(cb + lp) * KP;
      v8f acc = {};
#pragma unroll
      for (int ch = 0; ch < KP / 32; ch++) {
        v16h bm = load_b_chunk(Brow, ch * 32, half);
        acc = __builtin_amdgcn_wmma_f32_16x16x32_f16(false, a[ch], false, bm, (short)0, acc, false, false);
      }
      float sqj = sqb[cb + lp];
      F8 u; u.v = acc;
#pragma unroll
      for (int j = 0; j < 8; j++) {
        int M = j + half * 8;
        sdist[M][(t << 4) + lp] = sqi[j] + sqj - 2.f * u.f[j];
      }
    }
    __syncthreads();

    // ---- phase 2: merge chunk into running top-20 (wave owns rows 2*wid, 2*wid+1) ----
    for (int rr = 0; rr < 2; rr++) {
      int r = wid * 2 + rr;
      if (lane < 20) sdist[r][C + lane] = srunv[r][lane];
      asm volatile("" ::: "memory");
      for (int it = 0; it < 20; it++) {
        float bv = 3.2e38f; int bre = 0x7fffffff, bjj = 0;
        for (int jj = lane; jj < C + 20; jj += 32) {
          float d = sdist[r][jj];
          int real = (jj < C) ? (cc + jj) : sruni[r][jj - C];
          if (d < bv || (d == bv && real < bre)) { bv = d; bre = real; bjj = jj; }
        }
        for (int off = 16; off; off >>= 1) {
          float ov = __shfl_xor(bv, off, 32);
          int ore = __shfl_xor(bre, off, 32);
          int ojj = __shfl_xor(bjj, off, 32);
          if (ov < bv || (ov == bv && ore < bre)) { bv = ov; bre = ore; bjj = ojj; }
        }
        if (lane == 0) { sdist[r][bjj] = 3.3e38f; snewv[r][it] = bv; snewi[r][it] = bre; }
        asm volatile("" ::: "memory");
      }
      if (lane < 20) { srunv[r][lane] = snewv[r][lane]; sruni[r][lane] = snewi[r][lane]; }
      asm volatile("" ::: "memory");
    }
    __syncthreads();
  }

  for (int rr = 0; rr < 2; rr++) {
    int r = wid * 2 + rr;
    if (lane < 20) idxout[(size_t)(b * N + rb + r) * 20 + lane] = sruni[r][lane];
  }
}

// ================= gather-max: out = U[n] + max_k V[idx[n][k]] =================
__global__ void gather_max_kernel(const float* __restrict__ U, const float* __restrict__ V,
                                  const int* __restrict__ idx, float* __restrict__ Y,
                                  _Float16* __restrict__ Yh, int N, int O) {
  size_t i = (size_t)blockIdx.x * 256 + threadIdx.x;
  int o = (int)(i % O);
  size_t n = i / O;
  int b = (int)(n / N);
  const int* id = idx + n * 20;
  float m = -3.4e38f;
  for (int k = 0; k < 20; k++) {
    float v = V[((size_t)b * N + id[k]) * O + o];
    m = m > v ? m : v;
  }
  float y = U[n * O + o] + m;
  Y[i] = y;
  if (Yh) Yh[i] = (_Float16)y;
}

__global__ void rowsq_kernel(const float* __restrict__ X, float* __restrict__ sq, int O) {
  int n = blockIdx.x * 256 + threadIdx.x;
  float s = 0.f;
  for (int o = 0; o < O; o++) { float v = X[(size_t)n * O + o]; s += v * v; }
  sq[n] = s;
}

// ================= batchnorm over rows =================
__global__ void bn_stats_kernel(const float* __restrict__ H, const float* __restrict__ g,
                                const float* __restrict__ be, float* __restrict__ scale,
                                float* __restrict__ shift, int Nn, int O) {
  int o = blockIdx.x;
  __shared__ float ss[256], ss2[256];
  float s = 0.f, s2 = 0.f;
  for (int n = threadIdx.x; n < Nn; n += 256) {
    float v = H[(size_t)n * O + o];
    s += v; s2 += v * v;
  }
  ss[threadIdx.x] = s; ss2[threadIdx.x] = s2;
  __syncthreads();
  for (int st = 128; st; st >>= 1) {
    if (threadIdx.x < st) { ss[threadIdx.x] += ss[threadIdx.x + st]; ss2[threadIdx.x] += ss2[threadIdx.x + st]; }
    __syncthreads();
  }
  if (!threadIdx.x) {
    float m = ss[0] / Nn;
    float v = ss2[0] / Nn - m * m;
    float sc = g[o] * rsqrtf(v + 1e-5f);
    scale[o] = sc;
    shift[o] = be[o] - m * sc;
  }
}

__global__ void bn_apply_kernel(const float* __restrict__ H, const float* __restrict__ scale,
                                const float* __restrict__ shift, float* __restrict__ Y,
                                _Float16* __restrict__ Yh, int O) {
  size_t i = (size_t)blockIdx.x * 256 + threadIdx.x;
  int o = (int)(i % O);
  float y = H[i] * scale[o] + shift[o];
  if (Y) Y[i] = y;
  if (Yh) Yh[i] = (_Float16)y;
}

// ================= global attention branch =================
__global__ void gmean_kernel(const float* __restrict__ G, float* __restrict__ gmean, int N) {
  int b = blockIdx.x >> 6, o = blockIdx.x & 63;
  __shared__ float ss[256];
  float s = 0.f;
  for (int n = threadIdx.x; n < N; n += 256) s += G[((size_t)b * N + n) * 64 + o];
  ss[threadIdx.x] = s;
  __syncthreads();
  for (int st = 128; st; st >>= 1) {
    if (threadIdx.x < st) ss[threadIdx.x] += ss[threadIdx.x + st];
    __syncthreads();
  }
  if (!threadIdx.x) gmean[b * 64 + o] = ss[0] / N;
}

__global__ void gw_kernel(const float* __restrict__ gmean, const float* __restrict__ w1,
                          const float* __restrict__ b1, const float* __restrict__ w2,
                          const float* __restrict__ b2, float* __restrict__ gw) {
  int t = threadIdx.x;
  int b = t >> 6, o = t & 63;
  float acc = b2[o];
  for (int j = 0; j < 32; j++) {
    float h = b1[j];
    for (int f = 0; f < 64; f++) h += gmean[b * 64 + f] * w1[f * 32 + j];
    h = h > 0.f ? h : 0.f;
    acc += h * w2[j * 64 + o];
  }
  gw[t] = 1.f / (1.f + expf(-acc));
}

// ================= build fused feature (f16) =================
__global__ void comb_kernel(const float* __restrict__ s1, const float* __restrict__ s2,
                            const float* __restrict__ s3, const float* __restrict__ G,
                            const float* __restrict__ lwlin, const float* __restrict__ gw,
                            _Float16* __restrict__ combh, int N) {
  size_t i = (size_t)blockIdx.x * 256 + threadIdx.x;
  int c = (int)(i & 511);
  size_t n = i >> 9;
  float v;
  if (c < 64) v = s1[n * 64 + c];
  else if (c < 192) v = s2[n * 128 + (c - 64)];
  else if (c < 448) v = s3[n * 256 + (c - 192)];
  else {
    int o = c - 448;
    int b = (int)(n / N);
    float lw = 1.f / (1.f + expf(-lwlin[n * 64 + o]));
    v = G[n * 64 + o] * (lw + gw[b * 64 + o]);
  }
  combh[i] = (_Float16)v;
}

// ================= ordinal head =================
__global__ void head_kernel(const float* __restrict__ h2, const float* __restrict__ w3,
                            const float* __restrict__ b3, const float* __restrict__ ordw,
                            const float* __restrict__ ordb, const float* __restrict__ ordcut,
                            float* __restrict__ out) {
  int n = blockIdx.x * 256 + threadIdx.x;
  float s = ordb[0];
  for (int c = 0; c < 5; c++) {
    float l = b3[c];
    for (int f = 0; f < 128; f++) l += h2[(size_t)n * 128 + f] * w3[f * 5 + c];
    s += l * ordw[c];
  }
  for (int c = 0; c < 4; c++) out[(size_t)n * 4 + c] = 1.f / (1.f + expf(-(s - ordcut[c])));
}

// =========================================================================================
extern "C" void kernel_launch(void* const* d_in, const int* in_sizes, int n_in,
                              void* d_out, int out_size, void* d_ws, size_t ws_size,
                              hipStream_t stream) {
  const float* x       = (const float*)d_in[0];
  const float* conv1_w = (const float*)d_in[2];  const float* conv1_b = (const float*)d_in[3];
  const float* conv2_w = (const float*)d_in[4];  const float* conv2_b = (const float*)d_in[5];
  const float* conv3_w = (const float*)d_in[6];  const float* conv3_b = (const float*)d_in[7];
  const float* ge_w1 = (const float*)d_in[8];   const float* ge_b1 = (const float*)d_in[9];
  const float* ge_g1 = (const float*)d_in[10];  const float* ge_be1 = (const float*)d_in[11];
  const float* ge_w2 = (const float*)d_in[12];  const float* ge_b2 = (const float*)d_in[13];
  const float* ge_g2 = (const float*)d_in[14];  const float* ge_be2 = (const float*)d_in[15];
  const float* la_w1 = (const float*)d_in[16];  const float* la_b1 = (const float*)d_in[17];
  const float* la_w2 = (const float*)d_in[18];  const float* la_b2 = (const float*)d_in[19];
  const float* ga_w1 = (const float*)d_in[20];  const float* ga_b1 = (const float*)d_in[21];
  const float* ga_w2 = (const float*)d_in[22];  const float* ga_b2 = (const float*)d_in[23];
  const float* fu_w1 = (const float*)d_in[24];  const float* fu_b1 = (const float*)d_in[25];
  const float* fu_g1 = (const float*)d_in[26];  const float* fu_be1 = (const float*)d_in[27];
  const float* fu_w2 = (const float*)d_in[28];  const float* fu_b2 = (const float*)d_in[29];
  const float* fu_g2 = (const float*)d_in[30];  const float* fu_be2 = (const float*)d_in[31];
  const float* fu_w3 = (const float*)d_in[32];  const float* fu_b3 = (const float*)d_in[33];
  const float* ord_w = (const float*)d_in[34];  const float* ord_b = (const float*)d_in[35];
  const float* ord_cut = (const float*)d_in[36];
  float* out = (float*)d_out;

  const int Nn = in_sizes[0] / 9;   // 16384
  const int B = 4;
  const int N = Nn / B;             // 4096

  // bump allocator over d_ws
  char* p = (char*)d_ws;
  auto alloc = [&](size_t bytes) -> void* {
    void* r = (void*)p;
    p += (bytes + 255) & ~(size_t)255;
    return r;
  };
  _Float16* xyzh  = (_Float16*)alloc((size_t)Nn * 32 * 2);
  _Float16* geomh = (_Float16*)alloc((size_t)Nn * 32 * 2);
  float* sqA = (float*)alloc((size_t)Nn * 4);
  float* sqB = (float*)alloc((size_t)Nn * 4);
  float* sqC = (float*)alloc((size_t)Nn * 4);
  int*   idx = (int*)alloc((size_t)Nn * 20 * 4);
  float* U   = (float*)alloc((size_t)Nn * 256 * 4);
  float* V   = (float*)alloc((size_t)Nn * 256 * 4);
  float* s1  = (float*)alloc((size_t)Nn * 64 * 4);
  _Float16* s1h = (_Float16*)alloc((size_t)Nn * 64 * 2);
  float* s2  = (float*)alloc((size_t)Nn * 128 * 4);
  _Float16* s2h = (_Float16*)alloc((size_t)Nn * 128 * 2);
  float* s3  = (float*)alloc((size_t)Nn * 256 * 4);
  _Float16* wd1t = (_Float16*)alloc(64 * 32 * 2);
  _Float16* wb1t = (_Float16*)alloc(64 * 32 * 2);
  _Float16* wd2t = (_Float16*)alloc(128 * 64 * 2);
  _Float16* wb2t = (_Float16*)alloc(128 * 64 * 2);
  _Float16* wd3t = (_Float16*)alloc(256 * 128 * 2);
  _Float16* wb3t = (_Float16*)alloc(256 * 128 * 2);
  _Float16* gw1t = (_Float16*)alloc(32 * 32 * 2);
  _Float16* gw2t = (_Float16*)alloc(64 * 32 * 2);
  _Float16* lw1t = (_Float16*)alloc(32 * 64 * 2);
  _Float16* lw2t = (_Float16*)alloc(64 * 32 * 2);
  _Float16* fw1t = (_Float16*)alloc((size_t)256 * 512 * 2);
  _Float16* fw2t = (_Float16*)alloc((size_t)128 * 256 * 2);
  float* g1raw = (float*)alloc((size_t)Nn * 32 * 4);
  _Float16* g1h = (_Float16*)alloc((size_t)Nn * 32 * 2);
  float* g2raw = (float*)alloc((size_t)Nn * 64 * 4);
  float* gbuf  = (float*)alloc((size_t)Nn * 64 * 4);
  _Float16* gh = (_Float16*)alloc((size_t)Nn * 64 * 2);
  _Float16* t1h = (_Float16*)alloc((size_t)Nn * 32 * 2);
  float* lwlin = (float*)alloc((size_t)Nn * 64 * 4);
  float* gmean = (float*)alloc(256 * 4);
  float* gwv   = (float*)alloc(256 * 4);
  _Float16* combh = (_Float16*)alloc((size_t)Nn * 512 * 2);
  float* h1raw = (float*)alloc((size_t)Nn * 256 * 4);
  _Float16* h1h = (_Float16*)alloc((size_t)Nn * 256 * 2);
  float* h2raw = (float*)alloc((size_t)Nn * 128 * 4);
  float* h2    = (float*)alloc((size_t)Nn * 128 * 4);
  float* scale = (float*)alloc(512 * 4);
  float* shift = (float*)alloc(512 * 4);

  auto gemm = [&](const _Float16* X, const _Float16* Wt, const float* bias,
                  float* Y, _Float16* Yh, int Kpad, int O, int relu) {
    int blocks = (Nn / 16) * (O / 16) / 8;
    gemm_kernel<<<blocks, 256, 0, stream>>>(X, Wt, bias, Y, Yh, Kpad, O, relu);
  };
  auto prepwt = [&](const float* w, _Float16* wt, int Kin, int O, int Kpad, int rowoff, int diffoff) {
    int total = O * Kpad;
    prep_wt_kernel<<<(total + 255) / 256, 256, 0, stream>>>(w, wt, Kin, O, Kpad, rowoff, diffoff);
  };

  // ---- prep ----
  prep_inputs_kernel<<<Nn / 256, 256, 0, stream>>>(x, xyzh, geomh, sqA);
  prepwt(conv1_w, wd1t, 3, 64, 32, 0, 3);
  prepwt(conv1_w, wb1t, 3, 64, 32, 3, -1);
  prepwt(conv2_w, wd2t, 64, 128, 64, 0, 64);
  prepwt(conv2_w, wb2t, 64, 128, 64, 64, -1);
  prepwt(conv3_w, wd3t, 128, 256, 128, 0, 128);
  prepwt(conv3_w, wb3t, 128, 256, 128, 128, -1);
  prepwt(ge_w1, gw1t, 6, 32, 32, 0, -1);
  prepwt(ge_w2, gw2t, 32, 64, 32, 0, -1);
  prepwt(la_w1, lw1t, 64, 32, 64, 0, -1);
  prepwt(la_w2, lw2t, 32, 64, 32, 0, -1);
  prepwt(fu_w1, fw1t, 512, 256, 512, 0, -1);
  prepwt(fu_w2, fw2t, 256, 128, 256, 0, -1);

  // ---- conv1 ----
  knn_kernel<32><<<B * (N / 16), 256, 0, stream>>>(xyzh, sqA, idx, N);
  gemm(xyzh, wd1t, conv1_b, U, nullptr, 32, 64, 0);
  gemm(xyzh, wb1t, nullptr, V, nullptr, 32, 64, 0);
  gather_max_kernel<<<(size_t)Nn * 64 / 256, 256, 0, stream>>>(U, V, idx, s1, s1h, N, 64);
  rowsq_kernel<<<Nn / 256, 256, 0, stream>>>(s1, sqB, 64);

  // ---- conv2 ----
  knn_kernel<64><<<B * (N / 16), 256, 0, stream>>>(s1h, sqB, idx, N);
  gemm(s1h, wd2t, conv2_b, U, nullptr, 64, 128, 0);
  gemm(s1h, wb2t, nullptr, V, nullptr, 64, 128, 0);
  gather_max_kernel<<<(size_t)Nn * 128 / 256, 256, 0, stream>>>(U, V, idx, s2, s2h, N, 128);
  rowsq_kernel<<<Nn / 256, 256, 0, stream>>>(s2, sqC, 128);

  // ---- conv3 ----
  knn_kernel<128><<<B * (N / 16), 256, 0, stream>>>(s2h, sqC, idx, N);
  gemm(s2h, wd3t, conv3_b, U, nullptr, 128, 256, 0);
  gemm(s2h, wb3t, nullptr, V, nullptr, 128, 256, 0);
  gather_max_kernel<<<(size_t)Nn * 256 / 256, 256, 0, stream>>>(U, V, idx, s3, nullptr, N, 256);

  // ---- geometric branch ----
  gemm(geomh, gw1t, ge_b1, g1raw, nullptr, 32, 32, 1);
  bn_stats_kernel<<<32, 256, 0, stream>>>(g1raw, ge_g1, ge_be1, scale, shift, Nn, 32);
  bn_apply_kernel<<<(size_t)Nn * 32 / 256, 256, 0, stream>>>(g1raw, scale, shift, nullptr, g1h, 32);
  gemm(g1h, gw2t, ge_b2, g2raw, nullptr, 32, 64, 1);
  bn_stats_kernel<<<64, 256, 0, stream>>>(g2raw, ge_g2, ge_be2, scale, shift, Nn, 64);
  bn_apply_kernel<<<(size_t)Nn * 64 / 256, 256, 0, stream>>>(g2raw, scale, shift, gbuf, gh, 64);
  gemm(gh, lw1t, la_b1, nullptr, t1h, 64, 32, 1);
  gemm(t1h, lw2t, la_b2, lwlin, nullptr, 32, 64, 0);
  gmean_kernel<<<B * 64, 256, 0, stream>>>(gbuf, gmean, N);
  gw_kernel<<<1, 256, 0, stream>>>(gmean, ga_w1, ga_b1, ga_w2, ga_b2, gwv);

  // ---- fusion ----
  comb_kernel<<<(size_t)Nn * 512 / 256, 256, 0, stream>>>(s1, s2, s3, gbuf, lwlin, gwv, combh, N);
  gemm(combh, fw1t, fu_b1, h1raw, nullptr, 512, 256, 1);
  bn_stats_kernel<<<256, 256, 0, stream>>>(h1raw, fu_g1, fu_be1, scale, shift, Nn, 256);
  bn_apply_kernel<<<(size_t)Nn * 256 / 256, 256, 0, stream>>>(h1raw, scale, shift, nullptr, h1h, 256);
  gemm(h1h, fw2t, fu_b2, h2raw, nullptr, 256, 128, 1);
  bn_stats_kernel<<<128, 256, 0, stream>>>(h2raw, fu_g2, fu_be2, scale, shift, Nn, 128);
  bn_apply_kernel<<<(size_t)Nn * 128 / 256, 256, 0, stream>>>(h2raw, scale, shift, h2, nullptr, 128);

  // ---- head ----
  head_kernel<<<Nn / 256, 256, 0, stream>>>(h2, fu_w3, fu_b3, ord_w, ord_b, ord_cut, out);
}
